// DH_GNN_67757404062071
// MI455X (gfx1250) — compile-verified, matrix operand order checked by reading, not run on hardware
//
#include <hip/hip_runtime.h>
#include <hip/hip_bf16.h>
#include <stdint.h>

// ---------------------------------------------------------------------------
// Problem constants (from reference)
// ---------------------------------------------------------------------------
#define NNODES 2000
#define NPER   1000
#define NB     2
#define EDGES  64000
#define DIN    512
#define HDIM   256
#define NHEADS 8
#define DHEAD  32
#define K1SEED 2500

// ---------------------------------------------------------------------------
// WMMA bf16 helpers (CDNA5 wave32, V_WMMA_F32_16X16X32_BF16)
// All K dims used here are multiples of 32 and all N dims multiples of 64,
// so fragment loads are branch-free: row indices are clamped (OOB rows are
// either discarded at the predicated store, or neutralized by the softmax
// mask / p==0 in the attention kernel).
// ---------------------------------------------------------------------------
typedef __attribute__((ext_vector_type(16))) __bf16 v16bf;
typedef __attribute__((ext_vector_type(8)))  float  v8f;

// 16-bit A/B fragment K index for element j of the per-lane v16 (ISA 7.12.2):
// lanes 0-15 hold K in {half*8+0..7} U {16+half*8+0..7} with half = lane>>4
__device__ __forceinline__ int frag_k(int j, int half) {
  return (j < 8) ? (half * 8 + j) : (16 + half * 8 + (j - 8));
}

// A fragment: rows (row0..row0+15) x K (k0..k0+31), row-major [rows x ld].
// Row index clamped; K range must be fully in-bounds (K multiple of 32).
// 4x float4 loads per lane (K-contiguous halves of the fragment).
__device__ __forceinline__ v16bf load_frag_A(const float* __restrict__ A, int ld,
                                             int row0, int k0, int row_lim) {
  int lane = threadIdx.x & 31;
  int r = row0 + (lane & 15);
  r = min(r, row_lim - 1);
  int half = lane >> 4;
  const float* p = A + (size_t)r * ld + (k0 + half * 8);
  float4 x0 = *(const float4*)(p);
  float4 x1 = *(const float4*)(p + 4);
  float4 x2 = *(const float4*)(p + 16);
  float4 x3 = *(const float4*)(p + 20);
  v16bf f;
  f[0] = (__bf16)x0.x;  f[1] = (__bf16)x0.y;  f[2] = (__bf16)x0.z;  f[3] = (__bf16)x0.w;
  f[4] = (__bf16)x1.x;  f[5] = (__bf16)x1.y;  f[6] = (__bf16)x1.z;  f[7] = (__bf16)x1.w;
  f[8] = (__bf16)x2.x;  f[9] = (__bf16)x2.y;  f[10] = (__bf16)x2.z; f[11] = (__bf16)x2.w;
  f[12] = (__bf16)x3.x; f[13] = (__bf16)x3.y; f[14] = (__bf16)x3.z; f[15] = (__bf16)x3.w;
  return f;
}

// B fragment: K (k0..k0+31) x cols (col0..col0+15), row-major [K x ld].
// K row index clamped to k_lim-1 (weights always exact; attention V uses the
// clamp for tail keys whose p is ~0). Lane-coalesced scalar loads.
__device__ __forceinline__ v16bf load_frag_B(const float* __restrict__ W, int ld,
                                             int k0, int col0, int k_lim) {
  int lane = threadIdx.x & 31;
  int c = col0 + (lane & 15);
  int kb = k0 + (lane >> 4) * 8;
  v16bf f;
#pragma unroll
  for (int j = 0; j < 8; ++j) {
    int ka = min(kb + j, k_lim - 1);
    int kc = min(kb + 16 + j, k_lim - 1);
    f[j]     = (__bf16)W[(size_t)ka * ld + c];
    f[j + 8] = (__bf16)W[(size_t)kc * ld + c];
  }
  return f;
}

__device__ __forceinline__ v8f wmma_bf16(v16bf a, v16bf b, v8f c) {
  return __builtin_amdgcn_wmma_f32_16x16x32_bf16(false, a, false, b, (short)0, c,
                                                 false, false);
}

// ---------------------------------------------------------------------------
// Generic GEMM: C[M,N] = epilogue(A[M,K] @ W[K,N]).
// One wave per 16x64 strip: A-frag loaded once per K-step, 4 B-frags, 4 WMMAs.
// Requires K % 32 == 0 and N % 64 == 0 (true for every call in this model).
// epilogue: optional +bias[n], optional +=C (addC), optional relu, optional *rowscale[m]
// ---------------------------------------------------------------------------
__global__ void gemm_wmma(const float* __restrict__ A, const float* __restrict__ W,
                          const float* __restrict__ bias, const float* __restrict__ rowscale,
                          float* __restrict__ C, int M, int N, int K, int addC, int relu) {
  int wave = threadIdx.x >> 5;
  int strip = blockIdx.x * 4 + wave;
  int nsn = N >> 6;                       // 64-col strips
  int ntm = (M + 15) >> 4;
  if (strip >= ntm * nsn) return;         // wave-uniform exit
  int row0 = (strip / nsn) * 16;
  int colb = (strip % nsn) * 64;

  v8f acc[4];
#pragma unroll
  for (int t = 0; t < 4; ++t) acc[t] = (v8f){0, 0, 0, 0, 0, 0, 0, 0};

  for (int k0 = 0; k0 < K; k0 += 32) {
    v16bf a = load_frag_A(A, K, row0, k0, M);
    // prefetch next A chunk (speculative; lowers to global_prefetch_b8)
    __builtin_prefetch(A + (size_t)(row0 + (threadIdx.x & 15)) * K + k0 + 32, 0, 1);
#pragma unroll
    for (int t = 0; t < 4; ++t) {
      v16bf b = load_frag_B(W, N, k0, colb + t * 16, K);
      acc[t] = wmma_bf16(a, b, acc[t]);
    }
  }

  int lane = threadIdx.x & 31, l15 = lane & 15;
  int mbase = row0 + (lane >> 4) * 8;
#pragma unroll
  for (int t = 0; t < 4; ++t) {
    int n = colb + t * 16 + l15;          // always < N (N % 64 == 0)
    float bv = (bias != nullptr) ? bias[n] : 0.0f;
#pragma unroll
    for (int r = 0; r < 8; ++r) {
      int m = mbase + r;
      if (m < M) {
        float v = acc[t][r] + bv;
        if (addC) v += C[(size_t)m * N + n];
        if (relu) v = fmaxf(v, 0.0f);
        if (rowscale != nullptr) v *= rowscale[m];
        C[(size_t)m * N + n] = v;
      }
    }
  }
}

// ---------------------------------------------------------------------------
// Flash attention for GMT MAB: one wave per (batch, head, 16-query tile).
//   O[b,m,h*32+d] = Q[b,m,h*32+d] + softmax(Q Kt / 16) V   (mask all-true here)
// Streams 32 keys per step: 2x QK^T WMMA -> online softmax (shuffle reductions
// within 16-lane halves; accumulator row layout m = (lane>>4)*8 + r matches the
// per-row rescale exactly) -> P through LDS into an A-frag -> 2x PV WMMA.
// Clamped loads: OOB key rows get score -1e30 (=> p ~ 0), OOB query rows are
// never stored, so the streaming loop is completely branch-free.
// ---------------------------------------------------------------------------
__global__ void attn_flash(const float* __restrict__ Q, long long qbstride,
                           const float* __restrict__ Kd, const float* __restrict__ Vd,
                           float* __restrict__ O, int Lq, int Lk, float scale) {
  int tilesq = (Lq + 15) >> 4;
  int id = blockIdx.x;
  int tq = id % tilesq; id /= tilesq;
  int hd = id % NHEADS;
  int b  = id / NHEADS;
  const float* qp = Q  + (size_t)b * (size_t)qbstride;
  const float* kp = Kd + (size_t)b * (size_t)Lk * HDIM;
  const float* vp = Vd + (size_t)b * (size_t)Lk * HDIM;
  int col0 = hd * DHEAD;
  int lane = threadIdx.x, half = lane >> 4, l15 = lane & 15;

  v16bf qa = load_frag_A(qp, HDIM, tq * 16, col0, Lq);   // fixed Q tile

  float mx[8], sm[8];
  v8f o0 = {0, 0, 0, 0, 0, 0, 0, 0}, o1 = {0, 0, 0, 0, 0, 0, 0, 0};
#pragma unroll
  for (int r = 0; r < 8; ++r) { mx[r] = -1e30f; sm[r] = 0.0f; }

  __shared__ __bf16 pbuf[16 * 32];   // 1KB probability tile

  for (int kb = 0; kb < Lk; kb += 32) {
    v8f z = {0, 0, 0, 0, 0, 0, 0, 0};
    // QK^T: keys as B-frags (key row -> column); same lane layout as an A load.
    v16bf kf0 = load_frag_A(kp, HDIM, kb,      col0, Lk);
    v16bf kf1 = load_frag_A(kp, HDIM, kb + 16, col0, Lk);
    v8f s0 = wmma_bf16(qa, kf0, z);
    v8f s1 = wmma_bf16(qa, kf1, z);

    int key0 = kb + l15, key1 = kb + 16 + l15;
    float a0 = (key0 < Lk) ? 0.0f : -1e30f;     // tail-key mask
    float a1 = (key1 < Lk) ? 0.0f : -1e30f;
    float t0[8], t1[8], tm[8];
#pragma unroll
    for (int r = 0; r < 8; ++r) {
      t0[r] = s0[r] * scale + a0;
      t1[r] = s1[r] * scale + a1;
      tm[r] = fmaxf(t0[r], t1[r]);
    }
#pragma unroll
    for (int d = 1; d < 16; d <<= 1)
#pragma unroll
      for (int r = 0; r < 8; ++r) tm[r] = fmaxf(tm[r], __shfl_xor(tm[r], d, 32));

    float p0[8], p1[8], rs[8], corr[8];
#pragma unroll
    for (int r = 0; r < 8; ++r) {
      float nm = fmaxf(mx[r], tm[r]);
      corr[r] = __expf(mx[r] - nm);
      mx[r] = nm;
      p0[r] = __expf(t0[r] - nm);
      p1[r] = __expf(t1[r] - nm);
      rs[r] = p0[r] + p1[r];
    }
#pragma unroll
    for (int d = 1; d < 16; d <<= 1)
#pragma unroll
      for (int r = 0; r < 8; ++r) rs[r] += __shfl_xor(rs[r], d, 32);
#pragma unroll
    for (int r = 0; r < 8; ++r) {
      sm[r] = sm[r] * corr[r] + rs[r];
      o0[r] *= corr[r];
      o1[r] *= corr[r];
    }

    // reshape P (D layout: row=half*8+r, col=key) -> A-frag via LDS
    __syncthreads();
#pragma unroll
    for (int r = 0; r < 8; ++r) {
      pbuf[(half * 8 + r) * 32 + l15]      = (__bf16)p0[r];
      pbuf[(half * 8 + r) * 32 + 16 + l15] = (__bf16)p1[r];
    }
    __syncthreads();
    v16bf pa;
#pragma unroll
    for (int j = 0; j < 16; ++j) pa[j] = pbuf[l15 * 32 + frag_k(j, half)];

    v16bf vb0 = load_frag_B(vp, HDIM, kb, col0,      Lk);
    v16bf vb1 = load_frag_B(vp, HDIM, kb, col0 + 16, Lk);
    o0 = wmma_bf16(pa, vb0, o0);
    o1 = wmma_bf16(pa, vb1, o1);
  }

#pragma unroll
  for (int r = 0; r < 8; ++r) {
    int m = tq * 16 + half * 8 + r;
    if (m < Lq) {
      float inv = 1.0f / sm[r];
      int d0 = col0 + l15, d1 = d0 + 16;
      size_t base = ((size_t)b * Lq + m) * HDIM;
      O[base + d0] = qp[(size_t)m * HDIM + d0] + o0[r] * inv;   // q residual
      O[base + d1] = qp[(size_t)m * HDIM + d1] + o1[r] * inv;
    }
  }
}

// ---------------------------------------------------------------------------
// Graph / elementwise kernels
// ---------------------------------------------------------------------------
__global__ void fill_kernel(float* p, size_t n, float v) {
  size_t i = (size_t)blockIdx.x * blockDim.x + threadIdx.x;
  if (i < n) p[i] = v;
}

__global__ void count_deg(const int* __restrict__ dst, float* cnt, int e) {
  int i = blockIdx.x * blockDim.x + threadIdx.x;
  if (i < e) atomicAdd(&cnt[dst[i]], 1.0f);
}

__global__ void make_dis(const float* __restrict__ cnt, float* dis, int n) {
  int i = blockIdx.x * blockDim.x + threadIdx.x;
  if (i < n) dis[i] = rsqrtf(cnt[i] + 1.0f);    // deg = indeg + self-loop >= 1
}

// agg[dst] += x[src]  (one block per edge, feature-parallel)
__global__ void edge_gather_add(const float* __restrict__ x, const int* __restrict__ src,
                                const int* __restrict__ dst, float* agg, int dim) {
  int e = blockIdx.x;
  int s = src[e], d = dst[e];
  for (int f = threadIdx.x; f < dim; f += blockDim.x)
    atomicAdd(&agg[(size_t)d * dim + f], x[(size_t)s * dim + f]);
}

__global__ void divide_mean(float* agg, const float* __restrict__ cnt, int n, int dim) {
  size_t i = (size_t)blockIdx.x * blockDim.x + threadIdx.x;
  if (i < (size_t)n * dim) agg[i] /= fmaxf(cnt[i / dim], 1.0f);
}

// out[n] = bias + dis[n]^2 * h[n]   (self-loop term, init)
__global__ void gcn_init(float* out, const float* __restrict__ h, const float* __restrict__ bias,
                         const float* __restrict__ dis, int n, int dim) {
  size_t i = (size_t)blockIdx.x * blockDim.x + threadIdx.x;
  if (i < (size_t)n * dim) {
    int r = (int)(i / dim), c = (int)(i % dim);
    out[i] = bias[c] + dis[r] * dis[r] * h[i];
  }
}

__global__ void gcn_edge(const float* __restrict__ h, const int* __restrict__ src,
                         const int* __restrict__ dst, const float* __restrict__ dis,
                         float* out, int dim) {
  int e = blockIdx.x;
  int s = src[e], d = dst[e];
  float w = dis[s] * dis[d];
  for (int f = threadIdx.x; f < dim; f += blockDim.x)
    atomicAdd(&out[(size_t)d * dim + f], w * h[(size_t)s * dim + f]);
}

__global__ void add_inplace(float* a, const float* __restrict__ b, size_t n) {
  size_t i = (size_t)blockIdx.x * blockDim.x + threadIdx.x;
  if (i < n) a[i] += b[i];
}

__global__ void copy_cols(float* dst, int ldd, int col0, const float* __restrict__ src,
                          int lds, int cols, int rows) {
  size_t i = (size_t)blockIdx.x * blockDim.x + threadIdx.x;
  if (i < (size_t)rows * cols) {
    int r = (int)(i / cols), c = (int)(i % cols);
    dst[(size_t)r * ldd + col0 + c] = src[(size_t)r * lds + c];
  }
}

__global__ void layernorm256(float* X, const float* __restrict__ g, const float* __restrict__ b) {
  __shared__ float red[256];
  int row = blockIdx.x, t = threadIdx.x;
  float v = X[(size_t)row * 256 + t];
  red[t] = v; __syncthreads();
  for (int s = 128; s > 0; s >>= 1) { if (t < s) red[t] += red[t + s]; __syncthreads(); }
  float mean = red[0] * (1.0f / 256.0f);
  __syncthreads();
  float d = v - mean;
  red[t] = d * d; __syncthreads();
  for (int s = 128; s > 0; s >>= 1) { if (t < s) red[t] += red[t + s]; __syncthreads(); }
  float var = red[0] * (1.0f / 256.0f);
  X[(size_t)row * 256 + t] = d * rsqrtf(var + 1e-5f) * g[t] + b[t];
}

__global__ void cosine128(const float* __restrict__ A, const float* __restrict__ Bm, float* out) {
  __shared__ float sd[128], sa[128], sb[128];
  int row = blockIdx.x, t = threadIdx.x;
  float a = A[(size_t)row * 128 + t], b = Bm[(size_t)row * 128 + t];
  sd[t] = a * b; sa[t] = a * a; sb[t] = b * b; __syncthreads();
  for (int s = 64; s > 0; s >>= 1) {
    if (t < s) { sd[t] += sd[t + s]; sa[t] += sa[t + s]; sb[t] += sb[t + s]; }
    __syncthreads();
  }
  if (t == 0) {
    float na = fmaxf(sqrtf(sa[0]), 1e-8f), nb = fmaxf(sqrtf(sb[0]), 1e-8f);
    out[row] = sd[0] / (na * nb);
  }
}

__device__ __forceinline__ float u01hash(unsigned x) {
  x ^= x >> 16; x *= 0x7feb352du; x ^= x >> 15; x *= 0x846ca68bu; x ^= x >> 16;
  return (float)(x >> 8) * (1.0f / 16777216.0f);
}

// x += sign * sign(x) * l2_normalize(U[0,1)) * 0.3  (hash PRNG stand-in for threefry)
__global__ void inject_noise(float* X, int dim, float sgn, unsigned salt) {
  __shared__ float red[256];
  int row = blockIdx.x, t = threadIdx.x;
  float ss = 0.0f;
  for (int f = t; f < dim; f += 256) {
    float r = u01hash((unsigned)(row * dim + f) * 2654435761u + salt * 0x9E3779B9u + 0x1234567u);
    ss += r * r;
  }
  red[t] = ss; __syncthreads();
  for (int s = 128; s > 0; s >>= 1) { if (t < s) red[t] += red[t + s]; __syncthreads(); }
  float inv = 0.3f * rsqrtf(red[0] + 1e-20f);
  for (int f = t; f < dim; f += 256) {
    float r = u01hash((unsigned)(row * dim + f) * 2654435761u + salt * 0x9E3779B9u + 0x1234567u);
    float xv = X[(size_t)row * dim + f];
    float s2 = (xv > 0.0f) ? 1.0f : ((xv < 0.0f) ? -1.0f : 0.0f);
    X[(size_t)row * dim + f] = xv + sgn * s2 * r * inv;
  }
}

// ---------------------------------------------------------------------------
// Host orchestration
// ---------------------------------------------------------------------------
extern "C" void kernel_launch(void* const* d_in, const int* in_sizes, int n_in,
                              void* d_out, int out_size, void* d_ws, size_t ws_size,
                              hipStream_t stream) {
  (void)in_sizes; (void)n_in; (void)out_size; (void)ws_size;
  int ix = 0;
  auto nextf = [&]() -> const float* { return (const float*)d_in[ix++]; };

  // ---- input mapping: top-level dict insertion order {x, params, edge_index, batch},
  //      params flattened jax-style (dict keys sorted; uppercase < lowercase) ----
  const float* x = nextf();                                   // [2000,512]
  struct Ad { const float *b, *w; };
  Ad ad_Mcold{nextf(), nextf()};                              // 'ad_Mcold' b,w
  Ad ad_Mhot {nextf(), nextf()};
  Ad ad_coldM{nextf(), nextf()};
  Ad ad_hotM {nextf(), nextf()};
  const float* S1 = nextf();                                  // gmt.S1 [2500,256]
  const float* S3 = nextf();                                  // gmt.S3 [1,256]
  const float* lin1_b = nextf(); const float* lin1_w = nextf();
  const float* lin2_b = nextf(); const float* lin2_w = nextf();
  struct Mab { const float *bk,*bo,*bq,*bv,*ln0b,*ln0g,*ln1b,*ln1g,*wk,*wo,*wq,*wv; };
  auto readMab = [&]() { Mab m;
    m.bk = nextf(); m.bo = nextf(); m.bq = nextf(); m.bv = nextf();
    m.ln0b = nextf(); m.ln0g = nextf(); m.ln1b = nextf(); m.ln1g = nextf();
    m.wk = nextf(); m.wo = nextf(); m.wq = nextf(); m.wv = nextf(); return m; };
  Mab p1 = readMab(), p2 = readMab(), p3 = readMab();
  struct Sg { const float *bl,*wl,*wr; };
  auto readSg = [&]() { Sg s; s.bl = nextf(); s.wl = nextf(); s.wr = nextf(); return s; };
  Sg sg1 = readSg(), sg1_cold = readSg(), sg1_hot = readSg();
  Sg sg2 = readSg(), sg2_cold = readSg(), sg2_hot = readSg();
  Sg sg3_1 = readSg(), sg3_2cold = readSg(), sg3_2hot = readSg();
  const int* ei = (const int*)d_in[ix++];
  const int* batchv = (const int*)d_in[ix++]; (void)batchv;   // contiguous by construction
  const int* src = ei;
  const int* dst = ei + EDGES;

  // ---- workspace (bump allocator; transient arena reused between phases) ----
  float* wsf = (float*)d_ws;
  size_t off = 0;
  auto alloc = [&](size_t n) { float* p = wsf + off; off += n; return p; };
  float* cnt = alloc(NNODES);
  float* dis = alloc(NNODES);
  float* m3 = alloc((size_t)NNODES * 512);
  float* h3 = alloc((size_t)NNODES * 512);
  float* c3 = alloc((size_t)NNODES * 512);
  size_t trans_base = off;

  // ---- small launch helpers ----
  auto zero = [&](float* p, size_t n) {
    fill_kernel<<<(int)((n + 255) / 256), 256, 0, stream>>>(p, n, 0.0f);
  };
  auto gemm = [&](const float* A, const float* W, const float* bias, const float* rsc,
                  float* C, int M, int N, int K, int addC, int relu) {
    int strips = ((M + 15) / 16) * (N / 64);
    gemm_wmma<<<(strips + 3) / 4, 128, 0, stream>>>(A, W, bias, rsc, C, M, N, K, addC, relu);
  };
  float* agg = nullptr;  // set per phase
  auto sage = [&](const float* xin, int din, int dout, const Sg& S, float* out) {
    zero(agg, (size_t)NNODES * din);
    edge_gather_add<<<EDGES, 256, 0, stream>>>(xin, src, dst, agg, din);
    divide_mean<<<(int)(((size_t)NNODES * din + 255) / 256), 256, 0, stream>>>(agg, cnt, NNODES, din);
    gemm(agg, S.wl, S.bl, nullptr, out, NNODES, dout, din, 0, 0);   // lin_l(mean)
    gemm(xin, S.wr, nullptr, nullptr, out, NNODES, dout, din, 1, 1); // += lin_r(x), relu
  };

  // ---- degrees (shared by all convs) ----
  zero(cnt, NNODES);
  count_deg<<<(EDGES + 255) / 256, 256, 0, stream>>>(dst, cnt, EDGES);
  make_dis<<<(NNODES + 255) / 256, 256, 0, stream>>>(cnt, dis, NNODES);

  // ================= Phase 1: dual hot/cold SAGE tower =================
  off = trans_base;
  float* m1 = alloc((size_t)NNODES * 64);
  float* h1 = alloc((size_t)NNODES * 64);
  float* c1 = alloc((size_t)NNODES * 64);
  float* m2 = alloc((size_t)NNODES * 128);
  float* h2 = alloc((size_t)NNODES * 128);
  float* c2 = alloc((size_t)NNODES * 128);
  float* cmh = alloc(NNODES);
  float* cmc = alloc(NNODES);
  float* m2_hot   = alloc((size_t)NNODES * 128);
  float* adahot2  = alloc((size_t)NNODES * 128);
  float* m2_cold  = alloc((size_t)NNODES * 128);
  float* adacold2 = alloc((size_t)NNODES * 128);
  float* m2cat = alloc((size_t)NNODES * 448);
  float* h2cat = alloc((size_t)NNODES * 320);
  float* c2cat = alloc((size_t)NNODES * 320);
  agg = alloc((size_t)NNODES * 512);

  sage(x, DIN, 64, sg1, m1);
  sage(x, DIN, 64, sg1_hot, h1);
  sage(x, DIN, 64, sg1_cold, c1);
  inject_noise<<<NNODES, 256, 0, stream>>>(h1, 64, +1.0f, 0u);
  inject_noise<<<NNODES, 256, 0, stream>>>(c1, 64, -1.0f, 1u);
  sage(m1, 64, 128, sg2, m2);
  sage(h1, 64, 128, sg2_hot, h2);
  sage(c1, 64, 128, sg2_cold, c2);
  cosine128<<<NNODES, 128, 0, stream>>>(m2, h2, cmh);
  cosine128<<<NNODES, 128, 0, stream>>>(m2, c2, cmc);
  gemm(m2, ad_hotM.w, ad_hotM.b, cmh, m2_hot,   NNODES, 128, 128, 0, 0);
  gemm(h2, ad_Mhot.w, ad_Mhot.b, cmh, adahot2,  NNODES, 128, 128, 0, 0);
  gemm(m2, ad_Mcold.w, ad_Mcold.b, cmc, m2_cold, NNODES, 128, 128, 0, 0);
  gemm(c2, ad_coldM.w, ad_coldM.b, cmc, adacold2, NNODES, 128, 128, 0, 0);
  // m2cat = [m2 | m1 | adahot2 | adacold2]
  auto cpc = [&](float* dstm, int ldd, int col0, const float* s, int c, int rws) {
    copy_cols<<<(int)(((size_t)rws * c + 255) / 256), 256, 0, stream>>>(dstm, ldd, col0, s, c, c, rws);
  };
  cpc(m2cat, 448, 0,   m2, 128, NNODES);
  cpc(m2cat, 448, 128, m1, 64,  NNODES);
  cpc(m2cat, 448, 192, adahot2, 128, NNODES);
  cpc(m2cat, 448, 320, adacold2, 128, NNODES);
  inject_noise<<<NNODES, 256, 0, stream>>>(h2, 128, +1.0f, 2u);   // h2n (in place, post-adapter)
  inject_noise<<<NNODES, 256, 0, stream>>>(c2, 128, -1.0f, 3u);
  cpc(h2cat, 320, 0,   h2, 128, NNODES);
  cpc(h2cat, 320, 128, h1, 64,  NNODES);
  cpc(h2cat, 320, 192, m2_hot, 128, NNODES);
  cpc(c2cat, 320, 0,   c2, 128, NNODES);
  cpc(c2cat, 320, 128, c1, 64,  NNODES);
  cpc(c2cat, 320, 192, m2_cold, 128, NNODES);
  sage(m2cat, 448, 512, sg3_1, m3);
  sage(h2cat, 320, 512, sg3_2hot, h3);
  sage(c2cat, 320, 512, sg3_2cold, c3);
  inject_noise<<<NNODES, 256, 0, stream>>>(h3, 512, +1.0f, 4u);
  inject_noise<<<NNODES, 256, 0, stream>>>(c3, 512, -1.0f, 5u);

  // ================= Phase 2: GMT pooling (x3) — reuse transient arena =====
  off = trans_base;
  float* g_h   = alloc((size_t)NNODES * HDIM);
  float* g_pre = alloc((size_t)NNODES * HDIM);
  float* g_Kd  = alloc((size_t)NNODES * HDIM);
  float* g_Vd  = alloc((size_t)NNODES * HDIM);
  float* g_q1  = alloc((size_t)K1SEED * HDIM);
  float* g_X   = alloc((size_t)NB * K1SEED * HDIM);
  float* g_Xo  = alloc((size_t)NB * K1SEED * HDIM);
  float* g_Q2  = alloc((size_t)NB * K1SEED * HDIM);
  float* g_K2  = alloc((size_t)NB * K1SEED * HDIM);
  float* g_V2  = alloc((size_t)NB * K1SEED * HDIM);
  float* g_T   = alloc((size_t)NB * K1SEED * HDIM);
  float* g_q3  = alloc(HDIM);
  float* g_X3  = alloc((size_t)NB * HDIM);

  auto run_mab = [&](const float* Qp, long long qbs, const float* Kp, const float* Vp,
                     float* Op, int Lq, int Lk, const Mab& p) {
    int tq = (Lq + 15) / 16;
    attn_flash<<<NB * NHEADS * tq, 32, 0, stream>>>(Qp, qbs, Kp, Vp, Op, Lq, Lk, 1.0f / 16.0f);
    layernorm256<<<NB * Lq, 256, 0, stream>>>(Op, p.ln0g, p.ln0b);
    gemm(Op, p.wo, p.bo, nullptr, g_T, NB * Lq, HDIM, HDIM, 0, 1);      // relu(fc_o)
    add_inplace<<<(int)(((size_t)NB * Lq * HDIM + 255) / 256), 256, 0, stream>>>(
        Op, g_T, (size_t)NB * Lq * HDIM);
    layernorm256<<<NB * Lq, 256, 0, stream>>>(Op, p.ln1g, p.ln1b);
  };
  auto gcnconv = [&](const float* hin, const float* w, const float* bias, float* outp) {
    gemm(hin, w, nullptr, nullptr, g_pre, NNODES, HDIM, HDIM, 0, 0);
    gcn_init<<<(int)(((size_t)NNODES * HDIM + 255) / 256), 256, 0, stream>>>(
        outp, g_pre, bias, dis, NNODES, HDIM);
    gcn_edge<<<EDGES, 256, 0, stream>>>(g_pre, src, dst, dis, outp, HDIM);
  };
  auto gmt = [&](const float* xn, float* outp) {
    gemm(xn, lin1_w, lin1_b, nullptr, g_h, NNODES, HDIM, 512, 0, 0);    // lin1
    // --- GMPool_G: GCN keys/values, dense() is a pure reshape (contiguous batch) ---
    gcnconv(g_h, p1.wk, p1.bk, g_Kd);
    gcnconv(g_h, p1.wv, p1.bv, g_Vd);
    gemm(S1, p1.wq, p1.bq, nullptr, g_q1, K1SEED, HDIM, HDIM, 0, 0);    // shared across batch
    run_mab(g_q1, 0LL, g_Kd, g_Vd, g_X, K1SEED, NPER, p1);
    // --- SelfAtt ---
    gemm(g_X, p2.wq, p2.bq, nullptr, g_Q2, NB * K1SEED, HDIM, HDIM, 0, 0);
    gemm(g_X, p2.wk, p2.bk, nullptr, g_K2, NB * K1SEED, HDIM, HDIM, 0, 0);
    gemm(g_X, p2.wv, p2.bv, nullptr, g_V2, NB * K1SEED, HDIM, HDIM, 0, 0);
    run_mab(g_Q2, (long long)K1SEED * HDIM, g_K2, g_V2, g_Xo, K1SEED, K1SEED, p2);
    // --- GMPool_I (1 seed) ---
    gemm(S3, p3.wq, p3.bq, nullptr, g_q3, 1, HDIM, HDIM, 0, 0);
    gemm(g_Xo, p3.wk, p3.bk, nullptr, g_K2, NB * K1SEED, HDIM, HDIM, 0, 0);
    gemm(g_Xo, p3.wv, p3.bv, nullptr, g_V2, NB * K1SEED, HDIM, HDIM, 0, 0);
    run_mab(g_q3, 0LL, g_K2, g_V2, g_X3, 1, K1SEED, p3);
    gemm(g_X3, lin2_w, lin2_b, nullptr, outp, NB, 512, HDIM, 0, 0);     // [B,512]
  };

  float* outv = (float*)d_out;
  gmt(m3, outv);            // main  -> out[0:1024]
  gmt(h3, outv + 1024);     // alpha -> out[1024:2048]
  gmt(c3, outv + 2048);     // beta  -> out[2048:3072]
}
